// MultiHeadAttention_16106127360592
// MI455X (gfx1250) — matrix operand order repaired and checked by reading, not compiled, not run on hardware
//
#include <hip/hip_runtime.h>
#include <hip/hip_bf16.h>
#include <math.h>

typedef __attribute__((ext_vector_type(16))) _Float16 v16h;
typedef __attribute__((ext_vector_type(8)))  _Float16 v8h;
typedef __attribute__((ext_vector_type(8)))  float    v8f;
typedef __attribute__((ext_vector_type(4)))  int      v4i;

#define D_MODEL 224
#define N_HEADS 7
#define BATCH   8
#define SEQ     1024
#define M_ROWS  (BATCH*SEQ)              /* 8192 */
#define LOG2E   1.4426950408889634f
#define QK_SCALE 0.17677669529663687f    /* 1/sqrt(32) */

union V16 { v16h v; v8h h[2]; };
union P8  { v8h h; int i[4]; };

static __device__ __forceinline__ v8f wmma_f16(const V16& a, const V16& b, v8f c) {
  // D(16x16,f32) = A(16x32,f16) x B(32x16,f16) + C
  return __builtin_amdgcn_wmma_f32_16x16x32_f16(false, a.v, false, b.v,
                                                (short)0, c, false, false);
}

// ---- CDNA5 async global->LDS copy (ASYNCcnt-tracked), 16B per lane ---------
typedef __attribute__((address_space(1))) v4i* glb_v4i_ptr;
typedef __attribute__((address_space(3))) v4i* lds_v4i_ptr;

static __device__ __forceinline__ void async_copy16(const void* g, void* l) {
#if __has_builtin(__builtin_amdgcn_global_load_async_to_lds_b128)
  __builtin_amdgcn_global_load_async_to_lds_b128(
      (glb_v4i_ptr)(g), (lds_v4i_ptr)(l), 0, 0);
#else
  unsigned lofs = (unsigned)(unsigned long long)l;   // LDS byte offset
  asm volatile("global_load_async_to_lds_b128 %0, %1, off"
               :: "v"(lofs), "v"(g) : "memory");
#endif
}

static __device__ __forceinline__ void wait_async0() {
#if __has_builtin(__builtin_amdgcn_s_wait_asynccnt)
  __builtin_amdgcn_s_wait_asynccnt(0);
#else
  asm volatile("s_wait_asynccnt 0" ::: "memory");
#endif
}

// ---------------------------------------------------------------------------
// Projection: Y = X(f32)[8192x224] @ W(f32)[224x224], emitted as f16.
// MODE 0: Q  -> [b,h,S,32], pre-scaled by 1/sqrt(d_k)
// MODE 1: K  -> [b,h,S,32]
// MODE 2: V  -> [b,h,32,S]  (transposed so attention A-operands are contiguous)
// ---------------------------------------------------------------------------
template <int MODE>
__global__ __launch_bounds__(256) void proj_kernel(
    const float* __restrict__ X, const float* __restrict__ W,
    _Float16* __restrict__ out)
{
  __shared__ __align__(32) _Float16 wt[D_MODEL * 32];  // [n][k'] transposed panel

  const int tid   = threadIdx.x;
  const int lane  = tid & 31;
  const int wv    = tid >> 5;
  const int wtile = blockIdx.x * 8 + wv;          // 896 blocks * 8 waves = 7168 tiles
  const int m0 = (wtile / 14) * 16;               // 512 row tiles
  const int n0 = (wtile % 14) * 16;               // 14 col tiles
  const int r   = lane & 15;
  const int hlf = lane >> 4;

  v8f acc = {};
  for (int k0 = 0; k0 < D_MODEL; k0 += 32) {
    // stage W[k0..k0+32][0..224] transposed as f16 (coalesced over columns)
    if (tid < D_MODEL) {
      #pragma unroll 8
      for (int kk = 0; kk < 32; ++kk)
        wt[tid * 32 + kk] = (_Float16)W[(size_t)(k0 + kk) * D_MODEL + tid];
    }
    __syncthreads();

    V16 a, b;
    const float* xr = X + (size_t)(m0 + r) * D_MODEL + k0;
    #pragma unroll
    for (int j = 0; j < 8; ++j) {
      a.h[0][j] = (_Float16)xr[hlf * 8 + j];
      a.h[1][j] = (_Float16)xr[16 + hlf * 8 + j];
    }
    const _Float16* wr = &wt[(n0 + r) * 32];
    b.h[0] = *(const v8h*)(wr + hlf * 8);
    b.h[1] = *(const v8h*)(wr + 16 + hlf * 8);
    acc = wmma_f16(a, b, acc);
    __syncthreads();
  }

  const float sc = (MODE == 0) ? QK_SCALE : 1.0f;
  #pragma unroll
  for (int g = 0; g < 8; ++g) {
    const int m   = m0 + hlf * 8 + g;             // global row = b*SEQ + s
    const int b_i = m >> 10;
    const int s   = m & (SEQ - 1);
    const int col = n0 + r;
    const int head = col >> 5;
    const int d    = col & 31;
    const float val = acc[g] * sc;
    if (MODE == 2)
      out[((size_t)(b_i * N_HEADS + head) * 32 + d) * SEQ + s] = (_Float16)val;
    else
      out[((size_t)(b_i * N_HEADS + head) * SEQ + s) * 32 + d] = (_Float16)val;
  }
}

// ---------------------------------------------------------------------------
// Fused flash attention, TRANSPOSED formulation, 64 keys per iteration:
//   S^T  = K(16key x 32d) x q^T(32d x 16q)      -> q is the COLUMN dim
//   ctx^T = V^T(16vd x 32key) x P^T(32key x 16q)
// One workgroup = (b, head, 64 q rows); 4 waves, each owns 16 q columns.
// Each lane owns ONE q column -> per-lane scalar m/l/alpha; masked keys are
// handled by zeroing p (max over raw scores is a valid upper bound for the
// online-softmax shift). K/V/mask stream via async global->LDS, double-buffered.
// ---------------------------------------------------------------------------
__global__ __launch_bounds__(128) void attn_kernel(
    const _Float16* __restrict__ qw, const _Float16* __restrict__ kw,
    const _Float16* __restrict__ vtw, const unsigned char* __restrict__ mask,
    _Float16* __restrict__ ctx)
{
  __shared__ __align__(16) _Float16      lds_k[2][64 * 32];  // [key][d]
  __shared__ __align__(16) _Float16      lds_v[2][32 * 64];  // [d][key]
  __shared__ __align__(16) unsigned char lds_m[2][64 * 64];  // [qrow][key]

  const int tid  = threadIdx.x;
  const int lane = tid & 31;
  const int wv   = tid >> 5;
  const int bh   = blockIdx.x >> 4;          // 56 (b,h) pairs
  const int qblk = blockIdx.x & 15;          // 16 q blocks of 64 rows
  const int b    = bh / N_HEADS;
  const int h0   = bh % N_HEADS;
  const int m0w  = qblk * 64 + wv * 16;

  const _Float16* qbase = qw  + (size_t)bh * SEQ * 32;
  const _Float16* kbase = kw  + (size_t)bh * SEQ * 32;
  const _Float16* vbase = vtw + (size_t)bh * 32 * SEQ;
  const unsigned char* mbase = mask + (size_t)b * SEQ * SEQ + (size_t)(qblk * 64) * SEQ;

  const int n   = lane & 15;                 // this lane's q column
  const int hlf = lane >> 4;

  // stage: 2x16B chunks per tensor per thread (K:4KB, V:4KB, mask:4KB)
  auto stage = [&](int kb, int bf) {
    #pragma unroll
    for (int i = 0; i < 2; ++i) {
      const int c = tid + 128 * i;           // 0..255
      const int kkey = c >> 2, kpart = c & 3;
      async_copy16(kbase + (size_t)(kb * 64 + kkey) * 32 + kpart * 8,
                   &lds_k[bf][kkey * 32 + kpart * 8]);
      const int vd = c >> 3, vpart = c & 7;
      async_copy16(vbase + (size_t)vd * SEQ + kb * 64 + vpart * 8,
                   &lds_v[bf][vd * 64 + vpart * 8]);
      const int mrow = c >> 2, mpart = c & 3;
      async_copy16(mbase + (size_t)mrow * SEQ + kb * 64 + mpart * 16,
                   &lds_m[bf][mrow * 64 + mpart * 16]);
    }
  };

  // B operand q^T: lane holds column q=m0w+n, d = hlf*8+0..7 and 16+hlf*8+0..7
  V16 bq;
  {
    const _Float16* qr = qbase + (size_t)(m0w + n) * 32;
    bq.h[0] = *(const v8h*)(qr + hlf * 8);
    bq.h[1] = *(const v8h*)(qr + 16 + hlf * 8);
  }

  v8f accT0 = {}, accT1 = {};   // ctx^T: vd rows [0,16) / [16,32), q col = n
  float m_run = -__builtin_inff();
  float l_run = 0.f;

  const int NB = SEQ / 64;
  int buf = 0;
  stage(0, 0);
  wait_async0();
  __syncthreads();

  for (int kb = 0; kb < NB; ++kb) {
    if (kb + 1 < NB) stage(kb + 1, buf ^ 1);   // overlap next tile with compute

    // 4 score tiles: s[kt][g] = S^T[key=kt*16+hlf*8+g][q=n]
    v8f s[4];
    #pragma unroll
    for (int kt = 0; kt < 4; ++kt) {
      V16 ak;
      const _Float16* kp = &lds_k[buf][(kt * 16 + n) * 32];
      ak.h[0] = *(const v8h*)(kp + hlf * 8);
      ak.h[1] = *(const v8h*)(kp + 16 + hlf * 8);
      v8f zc = {};
      s[kt] = wmma_f16(ak, bq, zc);
    }

    // mask words: this lane's q row, 8 bytes per tile as 2x u32
    const unsigned char* mr = &lds_m[buf][(wv * 16 + n) * 64];
    unsigned mw[8];
    #pragma unroll
    for (int kt = 0; kt < 4; ++kt) {
      const unsigned* p = (const unsigned*)(mr + kt * 16 + hlf * 8);
      mw[2 * kt]     = p[0];
      mw[2 * kt + 1] = p[1];
    }

    // per-column max over RAW scores (valid upper bound incl. masked keys)
    float loc = s[0][0];
    #pragma unroll
    for (int kt = 0; kt < 4; ++kt)
      #pragma unroll
      for (int g = 0; g < 8; ++g) loc = fmaxf(loc, s[kt][g]);
    const float rmax  = fmaxf(loc, __shfl_xor(loc, 16, 32));
    const float mnew  = fmaxf(m_run, rmax);
    const float alpha = exp2f((m_run - mnew) * LOG2E);

    // probabilities: masked -> 0; pack f16 for the P^T operand; local sum
    P8 my[4];
    float lsum = 0.f;
    #pragma unroll
    for (int kt = 0; kt < 4; ++kt) {
      #pragma unroll
      for (int g = 0; g < 8; ++g) {
        const unsigned mb = (mw[2 * kt + (g >> 2)] >> (8 * (g & 3))) & 0xffu;
        const float e = exp2f((s[kt][g] - mnew) * LOG2E);
        const float p = mb ? 0.f : e;
        my[kt].h[g] = (_Float16)p;
        lsum += p;
      }
    }
    l_run = l_run * alpha + (lsum + __shfl_xor(lsum, 16, 32));
    m_run = mnew;

    #pragma unroll
    for (int g = 0; g < 8; ++g) { accT0[g] *= alpha; accT1[g] *= alpha; }

    // two 32-key P^T groups; each needs one v8h cross-half exchange
    #pragma unroll
    for (int g2 = 0; g2 < 2; ++g2) {
      P8 snd, rx;
      snd.h = hlf ? my[2 * g2].h : my[2 * g2 + 1].h;
      #pragma unroll
      for (int i = 0; i < 4; ++i) rx.i[i] = __shfl_xor(snd.i[i], 16, 32);
      V16 bp;
      if (hlf == 0) { bp.h[0] = my[2 * g2].h; bp.h[1] = rx.h; }
      else          { bp.h[0] = rx.h;         bp.h[1] = my[2 * g2 + 1].h; }

      V16 av0, av1;
      const _Float16* v0p = &lds_v[buf][n * 64 + g2 * 32];
      av0.h[0] = *(const v8h*)(v0p + hlf * 16);
      av0.h[1] = *(const v8h*)(v0p + hlf * 16 + 8);
      const _Float16* v1p = &lds_v[buf][(16 + n) * 64 + g2 * 32];
      av1.h[0] = *(const v8h*)(v1p + hlf * 16);
      av1.h[1] = *(const v8h*)(v1p + hlf * 16 + 8);
      accT0 = wmma_f16(av0, bp, accT0);
      accT1 = wmma_f16(av1, bp, accT1);
    }

    wait_async0();      // next tile landed (no-op on last iteration)
    __syncthreads();
    buf ^= 1;
  }

  // normalize and emit ctx (f16) [b, s, h*32+vd]; vd is contiguous per lane
  const float inv = 1.0f / l_run;
  v8h o0, o1;
  #pragma unroll
  for (int g = 0; g < 8; ++g) {
    o0[g] = (_Float16)(accT0[g] * inv);   // vd = hlf*8+g
    o1[g] = (_Float16)(accT1[g] * inv);   // vd = 16+hlf*8+g
  }
  _Float16* crow = ctx + (size_t)(b * SEQ + m0w + n) * D_MODEL + h0 * 32;
  *(v8h*)(crow + hlf * 8)      = o0;
  *(v8h*)(crow + 16 + hlf * 8) = o1;
}

// ---------------------------------------------------------------------------
// Output projection: out(f32) = ctx(f16)[8192x224] @ W_O(f32)[224x224]
// ---------------------------------------------------------------------------
__global__ __launch_bounds__(256) void oproj_kernel(
    const _Float16* __restrict__ ctx, const float* __restrict__ W,
    float* __restrict__ out)
{
  __shared__ __align__(32) _Float16 wt[D_MODEL * 32];

  const int tid   = threadIdx.x;
  const int lane  = tid & 31;
  const int wv    = tid >> 5;
  const int wtile = blockIdx.x * 8 + wv;
  const int m0 = (wtile / 14) * 16;
  const int n0 = (wtile % 14) * 16;
  const int r   = lane & 15;
  const int hlf = lane >> 4;

  v8f acc = {};
  for (int k0 = 0; k0 < D_MODEL; k0 += 32) {
    if (tid < D_MODEL) {
      #pragma unroll 8
      for (int kk = 0; kk < 32; ++kk)
        wt[tid * 32 + kk] = (_Float16)W[(size_t)(k0 + kk) * D_MODEL + tid];
    }
    __syncthreads();

    V16 a, b;
    const _Float16* xr = ctx + (size_t)(m0 + r) * D_MODEL + k0;
    a.h[0] = *(const v8h*)(xr + hlf * 8);
    a.h[1] = *(const v8h*)(xr + 16 + hlf * 8);
    const _Float16* wr = &wt[(n0 + r) * 32];
    b.h[0] = *(const v8h*)(wr + hlf * 8);
    b.h[1] = *(const v8h*)(wr + 16 + hlf * 8);
    acc = wmma_f16(a, b, acc);
    __syncthreads();
  }
  #pragma unroll
  for (int g = 0; g < 8; ++g)
    out[(size_t)(m0 + hlf * 8 + g) * D_MODEL + n0 + r] = acc[g];
}

// ---------------------------------------------------------------------------
extern "C" void kernel_launch(void* const* d_in, const int* in_sizes, int n_in,
                              void* d_out, int out_size, void* d_ws, size_t ws_size,
                              hipStream_t stream) {
  (void)in_sizes; (void)n_in; (void)out_size;
  const float* Q  = (const float*)d_in[0];
  const float* K  = (const float*)d_in[1];
  const float* V  = (const float*)d_in[2];
  const unsigned char* mask = (const unsigned char*)d_in[3];
  const float* W_Q = (const float*)d_in[4];
  const float* W_K = (const float*)d_in[5];
  const float* W_V = (const float*)d_in[6];
  const float* W_O = (const float*)d_in[7];
  float* out = (float*)d_out;

  const size_t N = (size_t)M_ROWS * D_MODEL;   // 1,835,008 elems per f16 buffer
  if (ws_size < 4 * N * sizeof(_Float16)) return;
  _Float16* qw  = (_Float16*)d_ws;             // [b,h,S,32], pre-scaled
  _Float16* kw  = qw  + N;                     // [b,h,S,32]
  _Float16* vtw = kw  + N;                     // [b,h,32,S] (transposed)
  _Float16* cx  = vtw + N;                     // [b,s,224]

  const dim3 gblk(896), gthr(256);
  proj_kernel<0><<<gblk, gthr, 0, stream>>>(Q, W_Q, qw);
  proj_kernel<1><<<gblk, gthr, 0, stream>>>(K, W_K, kw);
  proj_kernel<2><<<gblk, gthr, 0, stream>>>(V, W_V, vtw);
  attn_kernel<<<dim3(BATCH * N_HEADS * (SEQ / 64)), dim3(128), 0, stream>>>(
      qw, kw, vtw, mask, cx);
  oproj_kernel<<<gblk, gthr, 0, stream>>>(cx, W_O, out);
}
